// PhasedClassifier_89129161327086
// MI455X (gfx1250) — matrix operand order repaired and checked by reading, not compiled, not run on hardware
//
#include <hip/hip_runtime.h>
#include <hip/hip_bf16.h>
#include <cstdint>
#include <cstddef>

// Problem constants (match reference)
#define BB 256
#define TT 500
#define DD 3
#define UU 512
#define CC 10
#define GG 2048  // 4*U

static constexpr float R_ON   = 0.05f;
static constexpr float ALPHAF = 1e-3f;
static constexpr float LN_EPS = 1e-3f;

typedef __bf16 v8bf  __attribute__((ext_vector_type(8)));
typedef __bf16 v16bf __attribute__((ext_vector_type(16)));
typedef float  v8f   __attribute__((ext_vector_type(8)));

__device__ __forceinline__ v16bf cat8(v8bf lo, v8bf hi) {
  return __builtin_shufflevector(lo, hi, 0,1,2,3,4,5,6,7,8,9,10,11,12,13,14,15);
}

__device__ __forceinline__ float sigmoidf_(float x) {
  return 1.0f / (1.0f + expf(-x));
}

// ---------------------------------------------------------------------------
// Zero-initialize recurrent state + bf16 activation buffers.
// ---------------------------------------------------------------------------
__global__ void init_state(float* h0, float* c0, float* h1, float* c1,
                           __bf16* h0bf, __bf16* acat) {
  const int i = blockIdx.x * blockDim.x + threadIdx.x;
  if (i < BB * UU) {
    h0[i] = 0.f; c0[i] = 0.f; h1[i] = 0.f; c1[i] = 0.f;
    h0bf[i] = (__bf16)0.0f;
  }
  if (i < BB * 2 * UU) acat[i] = (__bf16)0.0f;
}

// ---------------------------------------------------------------------------
// Pack fp32 weight matrix [K x 2048] (row-major; rows >= K0 come from src1,
// i.e. a virtual concat [src0; src1]) into the v_wmma_f32_16x16x32_bf16
// B-fragment layout: for tile (kt,nt) lane L holds column n = nt*16+(L&15),
// rows kb..kb+15 where kb = kt*32 + (L>>4)*16, 16 bf16 contiguous (32 bytes).
// ---------------------------------------------------------------------------
__global__ void pack_b(const float* __restrict__ src0, const float* __restrict__ src1,
                       int K0, int K, __bf16* __restrict__ dst) {
  const int idx = blockIdx.x * blockDim.x + threadIdx.x;
  const int total = (K / 32) * 128 * 32;
  if (idx >= total) return;
  const int lane = idx & 31;
  const int nt   = (idx >> 5) & 127;
  const int kt   = idx >> 12;          // 128*32 == 4096 == 1<<12
  const int n    = nt * 16 + (lane & 15);
  const int kb   = kt * 32 + ((lane >> 4) << 4);
  __bf16* d = dst + (size_t)idx * 16;
  #pragma unroll
  for (int jj = 0; jj < 16; ++jj) {
    const int k = kb + jj;
    const float v = (k < K0) ? src0[(size_t)k * GG + n]
                             : src1[(size_t)(k - K0) * GG + n];
    d[jj] = (__bf16)v;
  }
}

// ---------------------------------------------------------------------------
// Fused PhasedLSTM step.
//   LAYER==0: z = h0@R0 (WMMA) + x_t@W0 (scalar, D=3) + b0
//   LAYER==1: z = [a0|h1]@[W1;R1] (WMMA) + b1, then classifier head+softmax
// Each block owns 16 batch rows (block-private: GEMM consumes last-step state
// for exactly these rows, pointwise rewrites the same rows).
// ---------------------------------------------------------------------------
template <int LAYER>
__global__ __launch_bounds__(1024)
void plstm_step(const __bf16* __restrict__ Abuf,   // A operand, row-major [B x KDIM] bf16
                const __bf16* __restrict__ Wp,     // packed B operand
                const float* __restrict__ bias,    // [2048]
                const float* __restrict__ xin,     // layer0: x [B,T,3]
                const float* __restrict__ W0,      // layer0: [3 x 2048]
                const float* __restrict__ times,   // [B,T]
                const float* __restrict__ tau,
                const float* __restrict__ sgate,
                const float* __restrict__ gamma,
                const float* __restrict__ beta,
                float* __restrict__ hbuf,          // [B x U] fp32 state
                float* __restrict__ cbuf,          // [B x U] fp32 state
                __bf16* __restrict__ hbf_dst,      // bf16 h for next-step GEMM A
                __bf16* __restrict__ acat_dst,     // layer0: a0 -> acat cols [0,U)
                const float* __restrict__ Wf,      // layer1: [U x C]
                const float* __restrict__ bfc,     // layer1: [C]
                float* __restrict__ out,           // [B,T,C]
                int t) {
  constexpr int KDIM       = (LAYER == 0) ? UU : 2 * UU;
  constexpr int KT         = KDIM / 32;
  constexpr int HBF_STRIDE = (LAYER == 0) ? UU : 2 * UU;
  constexpr int HBF_OFF    = (LAYER == 0) ? 0 : UU;

  extern __shared__ char smem[];
  __bf16* As     = (__bf16*)smem;                   // 16 * KDIM bf16
  float*  zsl    = (float*)(smem + 16 * KDIM * 2);  // 16 * 2048 f32
  float*  red    = zsl + 16 * GG;                   // 2048 f32 (sum | sumsq)
  float*  stats  = red + 2048;                      // 16 * 2 f32
  float*  a1sl   = stats + 32;                      // layer1: 16 * U f32
  float*  logits = a1sl + 16 * UU;                  // layer1: 16 * C f32

  const int tid  = threadIdx.x;
  const int m0   = blockIdx.x * 16;   // first batch row of this block
  const int lane = tid & 31;
  const int wave = tid >> 5;          // 32 waves

  // ---- stage A slab (16 rows x KDIM, contiguous in global) into LDS via
  //      CDNA5 async global->LDS DMA (tracked by ASYNCcnt) ----
  {
    const __bf16* gbase = Abuf + (size_t)m0 * KDIM;
    // low 32 bits of the generic shared-pointer = LDS byte offset
    const unsigned lds_base = (unsigned)(uintptr_t)(void*)As;
    constexpr int CH = (16 * KDIM) / 8;  // number of 16-byte chunks
    #pragma unroll
    for (int i = tid; i < CH; i += 1024) {
      const unsigned ldsa = lds_base + (unsigned)(i * 16);
      const unsigned voff = (unsigned)(i * 16);
      asm volatile("global_load_async_to_lds_b128 %0, %1, %2"
                   :
                   : "v"(ldsa), "v"(voff), "s"(gbase)
                   : "memory");
    }
    asm volatile("s_wait_asynccnt 0x0" ::: "memory");
  }
  __syncthreads();

  // ---- WMMA GEMM: z[16 x 2048] += A[16 x KDIM] * W[KDIM x 2048] ----
  // Software-pipelined: B fragments for k-tile kt+1 are loaded while the
  // four WMMAs of k-tile kt execute (8 b128 loads in flight per WMMA group).
  v8f acc0 = {}, acc1 = {}, acc2 = {}, acc3 = {};
  {
    const int arow = lane & 15;           // A: M = lane (mod 16)
    const int asel = (lane >> 4) << 4;    // A: K phase 0 / 16 per half-wave
    const int ntb  = wave << 2;           // this wave's 4 N-tiles
    constexpr size_t KSTRIDE = (size_t)128 * 32 * 16;  // elems per k-tile slab
    const __bf16* bbase = Wp + (((size_t)ntb * 32 + lane) << 4);

    // preload kt = 0 B fragments
    v16bf nb0 = cat8(*(const v8bf*)(bbase +    0), *(const v8bf*)(bbase +    8));
    v16bf nb1 = cat8(*(const v8bf*)(bbase +  512), *(const v8bf*)(bbase +  520));
    v16bf nb2 = cat8(*(const v8bf*)(bbase + 1024), *(const v8bf*)(bbase + 1032));
    v16bf nb3 = cat8(*(const v8bf*)(bbase + 1536), *(const v8bf*)(bbase + 1544));

    for (int kt = 0; kt < KT; ++kt) {
      // A fragment: contiguous k = 16*(lane/16) + s, same as packed B -> two ds_load_b128 per lane
      const __bf16* ap = As + arow * KDIM + (kt << 5) + asel;
      const v16bf afrag = cat8(*(const v8bf*)ap, *(const v8bf*)(ap + 8));

      const v16bf b0 = nb0, b1 = nb1, b2 = nb2, b3 = nb3;
      if (kt + 1 < KT) {
        const __bf16* bp = bbase + (size_t)(kt + 1) * KSTRIDE;
        if (kt + 2 < KT)
          __builtin_prefetch(bbase + (size_t)(kt + 2) * KSTRIDE, 0, 1);
        nb0 = cat8(*(const v8bf*)(bp +    0), *(const v8bf*)(bp +    8));
        nb1 = cat8(*(const v8bf*)(bp +  512), *(const v8bf*)(bp +  520));
        nb2 = cat8(*(const v8bf*)(bp + 1024), *(const v8bf*)(bp + 1032));
        nb3 = cat8(*(const v8bf*)(bp + 1536), *(const v8bf*)(bp + 1544));
      }

      acc0 = __builtin_amdgcn_wmma_f32_16x16x32_bf16(false, afrag, false, b0, (short)0, acc0, false, false);
      acc1 = __builtin_amdgcn_wmma_f32_16x16x32_bf16(false, afrag, false, b1, (short)0, acc1, false, false);
      acc2 = __builtin_amdgcn_wmma_f32_16x16x32_bf16(false, afrag, false, b2, (short)0, acc2, false, false);
      acc3 = __builtin_amdgcn_wmma_f32_16x16x32_bf16(false, afrag, false, b3, (short)0, acc3, false, false);
    }
    // scatter C/D (f32 16x16: VGPR r -> M = r + (lane>=16)*8, N = lane&15)
    const int mb = (lane >> 4) << 3;
    const int nc = lane & 15;
    #pragma unroll
    for (int r = 0; r < 8; ++r) {
      zsl[(mb + r) * GG + (ntb + 0) * 16 + nc] = acc0[r];
      zsl[(mb + r) * GG + (ntb + 1) * 16 + nc] = acc1[r];
      zsl[(mb + r) * GG + (ntb + 2) * 16 + nc] = acc2[r];
      zsl[(mb + r) * GG + (ntb + 3) * 16 + nc] = acc3[r];
    }
  }
  __syncthreads();

  // ---- pointwise: gates, time gate, state blend; 64 threads per row ----
  const int row = tid >> 6;
  const int lr  = tid & 63;
  const int gb  = m0 + row;
  float hb[8];
  {
    const float tcur = times[(size_t)gb * TT + t];
    float xv0 = 0.f, xv1 = 0.f, xv2 = 0.f;
    if (LAYER == 0) {
      const float* xp = xin + ((size_t)gb * TT + t) * DD;
      xv0 = xp[0]; xv1 = xp[1]; xv2 = xp[2];
    }
    float sum = 0.f, sq = 0.f;
    #pragma unroll
    for (int j = 0; j < 8; ++j) {
      const int u = lr + (j << 6);
      float zi = zsl[row * GG + u]            + bias[u];
      float zf = zsl[row * GG + u + UU]       + bias[u + UU];
      float zg = zsl[row * GG + u + 2 * UU]   + bias[u + 2 * UU];
      float zo = zsl[row * GG + u + 3 * UU]   + bias[u + 3 * UU];
      if (LAYER == 0) {
        zi += xv0 * W0[u]          + xv1 * W0[GG + u]          + xv2 * W0[2 * GG + u];
        zf += xv0 * W0[u + UU]     + xv1 * W0[GG + u + UU]     + xv2 * W0[2 * GG + u + UU];
        zg += xv0 * W0[u + 2*UU]   + xv1 * W0[GG + u + 2*UU]   + xv2 * W0[2 * GG + u + 2*UU];
        zo += xv0 * W0[u + 3*UU]   + xv1 * W0[GG + u + 3*UU]   + xv2 * W0[2 * GG + u + 3*UU];
      }
      const float ig = sigmoidf_(zi);
      const float fg = sigmoidf_(zf);
      const float og = sigmoidf_(zo);
      const float gg = tanhf(zg);
      const size_t si = (size_t)gb * UU + u;
      const float cp = cbuf[si];
      const float hp = hbuf[si];
      const float cn = fg * cp + ig * gg;
      const float hn = og * tanhf(cn);
      // phased time gate
      const float tauu = tau[u];
      float rr = fmodf(tcur - sgate[u], tauu);
      rr = (rr < 0.f) ? rr + tauu : rr;
      const float phi = rr / tauu;
      const float k = (phi < 0.5f * R_ON) ? (phi * (2.f / R_ON))
                    : ((phi < R_ON) ? (2.f - phi * (2.f / R_ON)) : (ALPHAF * phi));
      const float hbl = k * hn + (1.f - k) * hp;
      const float cbl = k * cn + (1.f - k) * cp;
      hbuf[si] = hbl;
      cbuf[si] = cbl;
      hbf_dst[(size_t)gb * HBF_STRIDE + HBF_OFF + u] = (__bf16)hbl;
      hb[j] = hbl;
      sum += hbl; sq += hbl * hbl;
    }
    red[tid] = sum;
    red[1024 + tid] = sq;
  }
  __syncthreads();

  // ---- LayerNorm stats per row ----
  if (tid < 16) {
    float s = 0.f, q = 0.f;
    #pragma unroll 8
    for (int j = 0; j < 64; ++j) { s += red[tid * 64 + j]; q += red[1024 + tid * 64 + j]; }
    const float mean = s * (1.f / (float)UU);
    const float var  = q * (1.f / (float)UU) - mean * mean;
    stats[tid * 2]     = mean;
    stats[tid * 2 + 1] = rsqrtf(var + LN_EPS);
  }
  __syncthreads();

  // ---- normalize ----
  {
    const float mean = stats[row * 2];
    const float rstd = stats[row * 2 + 1];
    #pragma unroll
    for (int j = 0; j < 8; ++j) {
      const int u = lr + (j << 6);
      const float a = (hb[j] - mean) * rstd * gamma[u] + beta[u];
      if (LAYER == 0) acat_dst[(size_t)gb * (2 * UU) + u] = (__bf16)a;
      else            a1sl[row * UU + u] = a;
    }
  }

  // ---- layer1: classifier head + softmax ----
  if (LAYER == 1) {
    __syncthreads();
    const int r  = wave >> 1;          // 2 waves per row
    const int c0 = (wave & 1) * 5;     // 5 classes per wave
    float p0 = 0.f, p1 = 0.f, p2 = 0.f, p3 = 0.f, p4 = 0.f;
    for (int u = lane; u < UU; u += 32) {
      const float av = a1sl[r * UU + u];
      const float* wf = Wf + (size_t)u * CC + c0;
      p0 += av * wf[0]; p1 += av * wf[1]; p2 += av * wf[2];
      p3 += av * wf[3]; p4 += av * wf[4];
    }
    #pragma unroll
    for (int off = 16; off > 0; off >>= 1) {
      p0 += __shfl_down(p0, off, 32);
      p1 += __shfl_down(p1, off, 32);
      p2 += __shfl_down(p2, off, 32);
      p3 += __shfl_down(p3, off, 32);
      p4 += __shfl_down(p4, off, 32);
    }
    if (lane == 0) {
      logits[r * CC + c0 + 0] = p0 + bfc[c0 + 0];
      logits[r * CC + c0 + 1] = p1 + bfc[c0 + 1];
      logits[r * CC + c0 + 2] = p2 + bfc[c0 + 2];
      logits[r * CC + c0 + 3] = p3 + bfc[c0 + 3];
      logits[r * CC + c0 + 4] = p4 + bfc[c0 + 4];
    }
    __syncthreads();
    if (tid < 16) {
      const int gbr = m0 + tid;
      float mx = logits[tid * CC];
      #pragma unroll
      for (int c2 = 1; c2 < CC; ++c2) mx = fmaxf(mx, logits[tid * CC + c2]);
      float e[CC];
      float se = 0.f;
      #pragma unroll
      for (int c2 = 0; c2 < CC; ++c2) { e[c2] = expf(logits[tid * CC + c2] - mx); se += e[c2]; }
      const float inv = 1.f / se;
      float* op = out + ((size_t)gbr * TT + t) * CC;
      #pragma unroll
      for (int c2 = 0; c2 < CC; ++c2) op[c2] = e[c2] * inv;
    }
  }
}

// ---------------------------------------------------------------------------
extern "C" void kernel_launch(void* const* d_in, const int* in_sizes, int n_in,
                              void* d_out, int out_size, void* d_ws, size_t ws_size,
                              hipStream_t stream) {
  const float* x     = (const float*)d_in[0];
  const float* times = (const float*)d_in[1];
  const float* W0    = (const float*)d_in[2];
  const float* R0    = (const float*)d_in[3];
  const float* b0    = (const float*)d_in[4];
  const float* tau0  = (const float*)d_in[5];
  const float* s0    = (const float*)d_in[6];
  const float* g0    = (const float*)d_in[7];
  const float* be0   = (const float*)d_in[8];
  const float* W1    = (const float*)d_in[9];
  const float* R1    = (const float*)d_in[10];
  const float* b1    = (const float*)d_in[11];
  const float* tau1  = (const float*)d_in[12];
  const float* s1    = (const float*)d_in[13];
  const float* g1    = (const float*)d_in[14];
  const float* be1   = (const float*)d_in[15];
  const float* Wf    = (const float*)d_in[16];
  const float* bfc   = (const float*)d_in[17];
  float* out = (float*)d_out;

  // workspace carve-out (256B aligned)
  uint8_t* ws = (uint8_t*)d_ws;
  size_t off = 0;
  auto carve = [&](size_t bytes) -> void* {
    void* p = ws + off;
    off = (off + bytes + 255) & ~(size_t)255;
    return p;
  };
  float*  h0   = (float*)carve((size_t)BB * UU * 4);
  float*  c0   = (float*)carve((size_t)BB * UU * 4);
  float*  h1   = (float*)carve((size_t)BB * UU * 4);
  float*  c1   = (float*)carve((size_t)BB * UU * 4);
  __bf16* h0bf = (__bf16*)carve((size_t)BB * UU * 2);
  __bf16* acat = (__bf16*)carve((size_t)BB * 2 * UU * 2);
  __bf16* Wp0  = (__bf16*)carve((size_t)UU * GG * 2);       // packed R0
  __bf16* Wp1  = (__bf16*)carve((size_t)2 * UU * GG * 2);   // packed [W1;R1]

  // zero recurrent state + bf16 activations
  {
    const int n = BB * 2 * UU;
    init_state<<<(n + 255) / 256, 256, 0, stream>>>(h0, c0, h1, c1, h0bf, acat);
  }
  // pack weights into WMMA B-fragment layout (bf16)
  {
    const int tot0 = (UU / 32) * 128 * 32;
    const int tot1 = (2 * UU / 32) * 128 * 32;
    pack_b<<<(tot0 + 255) / 256, 256, 0, stream>>>(R0, R0, UU, UU, Wp0);
    pack_b<<<(tot1 + 255) / 256, 256, 0, stream>>>(W1, R1, UU, 2 * UU, Wp1);
  }

  const size_t smem0 = (size_t)16 * UU * 2      + (size_t)16 * GG * 4 + 2048 * 4 + 32 * 4;
  const size_t smem1 = (size_t)16 * 2 * UU * 2  + (size_t)16 * GG * 4 + 2048 * 4 + 32 * 4
                     + (size_t)16 * UU * 4 + (size_t)16 * CC * 4;

  for (int t = 0; t < TT; ++t) {
    plstm_step<0><<<dim3(BB / 16), dim3(1024), smem0, stream>>>(
        h0bf, Wp0, b0, x, W0, times, tau0, s0, g0, be0,
        h0, c0, h0bf, acat, nullptr, nullptr, nullptr, t);
    plstm_step<1><<<dim3(BB / 16), dim3(1024), smem1, stream>>>(
        acat, Wp1, b1, nullptr, nullptr, times, tau1, s1, g1, be1,
        h1, c1, acat, nullptr, Wf, bfc, out, t);
  }
}
